// ToKmerLayer_2448131358925
// MI455X (gfx1250) — matrix operand order repaired and checked by reading, not compiled, not run on hardware
//
#include <hip/hip_runtime.h>

typedef __attribute__((ext_vector_type(16))) _Float16 v16h;
typedef __attribute__((ext_vector_type(8)))  float    v8f;
typedef __attribute__((ext_vector_type(4)))  float    v4f;

#define NT 8  // 16-position tiles processed per wave (B tiles amortized 8x)

// One wave32 owns NT consecutive 16-position tiles of one batch row.
// Per tile: A = 16x32 f16 (K = KK*4 live, zero-padded), 4x
// v_wmma_f32_16x16x32_f16 (feature groups of 16) reusing the same A; B tiles
// are loaded once per wave. f32 accumulation of one-hot products is exact, so
// (score > KK-1) matches the reference bit-for-bit.
template <int KK>
__global__ __launch_bounds__(256) void kmer_conv_wmma(
    const float* __restrict__ in,   // (B, L, 4) one-hot f32
    const float* __restrict__ ker,  // (KK, 4, 64) f32
    float* __restrict__ out,        // (B, Lout, 64) f32
    int L, int Lout)
{
    const int lane = threadIdx.x & 31;
    const int b    = blockIdx.y;
    const int t0   = (blockIdx.x * 8 + (threadIdx.x >> 5)) * NT;
    int p0 = t0 * 16;
    if (p0 >= Lout) return;                 // wave-uniform

    const int  n    = lane & 15;            // column within 16-feature group
    const int  m0   = (lane < 16) ? 0 : 8;  // D-row base for this half-wave
    const int  j0   = (lane < 16) ? 0 : 2;  // first conv tap for this half-wave
    const long base = (long)b * L;

    // ---- B tiles (once per wave): lane holds column N=n,
    // half e<8 = W[K=8*(lane/16)+e, g*16+n]; halfs 8-15 (K 16-31) stay zero. ----
    v16h bt[4];
    #pragma unroll
    for (int g = 0; g < 4; ++g) {
        #pragma unroll
        for (int e = 0; e < 16; ++e) bt[g][e] = (_Float16)0.0f;
    }
    {
        #pragma unroll
        for (int e = 0; e < 8; ++e) {
            const int kx = (lane >> 4) * 8 + e;
            if (kx < KK * 4) {
                #pragma unroll
                for (int g = 0; g < 4; ++g)
                    bt[g][e] = (_Float16)ker[kx * 64 + g * 16 + n];
            }
        }
    }

    constexpr float thr = (float)(KK - 1);

    // Fully-valid tiles for this wave, then at most one partial tile.
    const int remaining = Lout - p0;
    int full = remaining >> 4;
    if (full > NT) full = NT;

    const float* inp  = in  + (base + p0 + n + j0) * 4;
    float*       outp = out + ((long)b * Lout + p0 + m0) * 64 + n;

    for (int tt = 0; tt < full; ++tt) {
        // ---- A tile: ISA 16-bit A layout (16x32, wave32) ----
        // lanes 0-15 : halfs 0-7 = K 0-7   (rows p, p+1)
        // lanes 16-31: halfs 0-7 = K 8-15  (rows p+2, p+3 if KK==4)
        v16h a;
        #pragma unroll
        for (int e = 0; e < 16; ++e) a[e] = (_Float16)0.0f;
        {
            v4f x = *(const v4f*)(inp);
            a[0] = (_Float16)x.x; a[1] = (_Float16)x.y;
            a[2] = (_Float16)x.z; a[3] = (_Float16)x.w;
            if (j0 + 1 < KK) {
                v4f y = *(const v4f*)(inp + 4);
                a[4] = (_Float16)y.x; a[5] = (_Float16)y.y;
                a[6] = (_Float16)y.z; a[7] = (_Float16)y.w;
            }
        }

        #pragma unroll
        for (int g = 0; g < 4; ++g) {
            v8f c = {};
            v8f d = __builtin_amdgcn_wmma_f32_16x16x32_f16(
                false, a, false, bt[g], (short)0, c, false, false);
            float* orow = outp + g * 16;
            #pragma unroll
            for (int r = 0; r < 8; ++r)
                orow[r * 64] = (d[r] > thr) ? 1.0f : 0.0f;
        }

        inp  += 16 * 4;     // 16 input rows
        outp += 16 * 64;    // 16 output rows
    }

    // ---- Partial tail tile (at most one wave per batch reaches this) ----
    const int rem = remaining - (full << 4);
    if (full < NT && rem > 0) {
        const int pT = p0 + (full << 4);
        v16h a;
        #pragma unroll
        for (int e = 0; e < 16; ++e) a[e] = (_Float16)0.0f;
        {
            int r0 = pT + n + j0;     if (r0 > L - 1) r0 = L - 1;
            v4f x = *(const v4f*)(in + (base + r0) * 4);
            a[0] = (_Float16)x.x; a[1] = (_Float16)x.y;
            a[2] = (_Float16)x.z; a[3] = (_Float16)x.w;
            if (j0 + 1 < KK) {
                int r1 = pT + n + j0 + 1; if (r1 > L - 1) r1 = L - 1;
                v4f y = *(const v4f*)(in + (base + r1) * 4);
                a[4] = (_Float16)y.x; a[5] = (_Float16)y.y;
                a[6] = (_Float16)y.z; a[7] = (_Float16)y.w;
            }
        }

        #pragma unroll
        for (int g = 0; g < 4; ++g) {
            v8f c = {};
            v8f d = __builtin_amdgcn_wmma_f32_16x16x32_f16(
                false, a, false, bt[g], (short)0, c, false, false);
            float* orow = out + ((long)b * Lout + pT + m0) * 64 + g * 16 + n;
            #pragma unroll
            for (int r = 0; r < 8; ++r)
                if (pT + m0 + r < Lout)
                    orow[r * 64] = (d[r] > thr) ? 1.0f : 0.0f;
        }
    }
}

extern "C" void kernel_launch(void* const* d_in, const int* in_sizes, int n_in,
                              void* d_out, int out_size, void* d_ws, size_t ws_size,
                              hipStream_t stream) {
    const float* in  = (const float*)d_in[0];   // (B, L, 4) f32
    const float* ker = (const float*)d_in[1];   // (k, 4, 64) f32
    float*       out = (float*)d_out;           // (B, L-k+1, 64) f32

    const int F  = 64;
    const int kk = in_sizes[1] / (4 * F);       // 3 for this problem
    const int B  = 64;
    const int L  = in_sizes[0] / (B * 4);       // 16384
    const int Lout = L - kk + 1;                // 16382

    const int tiles = (Lout + 15) / 16;         // 16-position tiles per batch
    const int wavesPerBatch = (tiles + NT - 1) / NT;
    dim3 grid((wavesPerBatch + 7) / 8, B);      // 8 waves (256 thr) per block
    dim3 block(256);

    switch (kk) {
    case 2:
        kmer_conv_wmma<2><<<grid, block, 0, stream>>>(in, ker, out, L, Lout);
        break;
    case 4:
        kmer_conv_wmma<4><<<grid, block, 0, stream>>>(in, ker, out, L, Lout);
        break;
    default: // k == 3 per the reference setup
        kmer_conv_wmma<3><<<grid, block, 0, stream>>>(in, ker, out, L, Lout);
        break;
    }
}
